// SlidingWindowAttention_6992206758374
// MI455X (gfx1250) — matrix operand order repaired and checked by reading, not compiled, not run on hardware
//
#include <hip/hip_runtime.h>

typedef __bf16 bf16_t;
typedef __attribute__((ext_vector_type(16))) __bf16 v16bf;
typedef __attribute__((ext_vector_type(8)))  __bf16 v8bf;
typedef __attribute__((ext_vector_type(8)))  float   v8f;

#define DIMSZ  2048
#define SEQ    2048
#define NHEAD  16
#define HDIM   128
#define NBATCH 2
#define WIN    128   // masked iff j - i > 128

union ABfrag { v16bf v; v8bf h[2]; };

__device__ __forceinline__ v8f wmma_bf16(v16bf a, v16bf b, v8f c) {
  // (neg_a, A, neg_b, B, c_mod, C, reuse_a, reuse_b)
  return __builtin_amdgcn_wmma_f32_16x16x32_bf16(false, a, false, b, (short)0, c,
                                                 false, false);
}

// ---------------------------------------------------------------- convert
__global__ void cvt_kernel(const float* __restrict__ src,
                           bf16_t* __restrict__ dst, int n) {
  int i = blockIdx.x * blockDim.x + threadIdx.x;
  if (i < n) dst[i] = (bf16_t)src[i];
}

// ---------------------------------------------------------------- GEMM
// C[m,n] = sum_k A[m,k] * W[n,k]   (A: [M,2048] bf16 row-major, W: [2048,2048])
// MODE 0: bf16 out [B,H,S,D]; MODE 1: bf16 out [B,H,D,S]; MODE 2: f32 out [M,N]
#define MODE_QK  0
#define MODE_VT  1
#define MODE_F32 2

// Block: 256 threads = 8 waves as 2(M) x 4(N); wave tile 64x64 (4x4 WMMA tiles).
// Block tile 128(M) x 256(N), K-step 32.
template <int MODE>
__global__ __launch_bounds__(256)
void gemm_kernel(const bf16_t* __restrict__ A, const bf16_t* __restrict__ W,
                 void* __restrict__ out) {
  const int lane = threadIdx.x & 31;
  const int wv   = threadIdx.x >> 5;
  const int h    = lane >> 4;       // lane half
  const int ln   = lane & 15;
  const int wm   = wv >> 2;         // 2 waves along M
  const int wn   = wv & 3;          // 4 waves along N
  const int m0   = blockIdx.y * 128 + wm * 64;
  const int n0   = blockIdx.x * 256 + wn * 64;

  v8f acc[4][4] = {};

  for (int kk = 0; kk < DIMSZ; kk += 32) {
    ABfrag a[4];
#pragma unroll
    for (int mt = 0; mt < 4; ++mt) {
      const bf16_t* ap = A + (size_t)(m0 + mt * 16 + ln) * DIMSZ + kk;
      a[mt].h[0] = *(const v8bf*)(ap + 16 * h);       // K = kk+16h .. +7
      a[mt].h[1] = *(const v8bf*)(ap + 16 * h + 8);   // K = kk+16h+8 .. +7
    }
    v16bf bfr[4];
#pragma unroll
    for (int nt = 0; nt < 4; ++nt) {
      const bf16_t* wp = W + (size_t)(n0 + nt * 16 + ln) * DIMSZ + kk + 16 * h;
      bfr[nt] = *(const v16bf*)wp;
      __builtin_prefetch(wp + 64, 0, 3);              // stream next K tiles
    }
#pragma unroll
    for (int mt = 0; mt < 4; ++mt)
#pragma unroll
      for (int nt = 0; nt < 4; ++nt)
        acc[mt][nt] = wmma_bf16(a[mt].v, bfr[nt], acc[mt][nt]);
  }

#pragma unroll
  for (int mt = 0; mt < 4; ++mt) {
    const int mg0 = m0 + mt * 16 + 8 * h;            // 8 consecutive rows follow
    const int bb  = mg0 >> 11;
    const int ss0 = mg0 & (SEQ - 1);
#pragma unroll
    for (int nt = 0; nt < 4; ++nt) {
      const int ng = n0 + nt * 16 + ln;
      if (MODE == MODE_F32) {
        float* op = (float*)out + (size_t)mg0 * DIMSZ + ng;
#pragma unroll
        for (int r = 0; r < 8; ++r)
          op[(size_t)r * DIMSZ] = acc[mt][nt][r];
      } else {
        const int hh = ng >> 7, dd = ng & (HDIM - 1);
        if (MODE == MODE_QK) {
          bf16_t* op = (bf16_t*)out +
                       (((size_t)(bb * NHEAD + hh)) * SEQ + ss0) * HDIM + dd;
#pragma unroll
          for (int r = 0; r < 8; ++r)
            op[(size_t)r * HDIM] = (bf16_t)acc[mt][nt][r];
        } else {                                     // MODE_VT: contiguous in s
          v8bf pv;
#pragma unroll
          for (int r = 0; r < 8; ++r) pv[r] = (bf16_t)acc[mt][nt][r];
          *(v8bf*)((bf16_t*)out +
                   (((size_t)(bb * NHEAD + hh)) * HDIM + dd) * SEQ + ss0) = pv;
        }
      }
    }
  }
}

// ---------------------------------------------------------------- attention
// Q,K: [B,H,S,128] bf16; VT: [B,H,128,S] bf16; Out: [B*S, 2048] bf16
// 1 wave = 16 queries, flash-style over 32-key blocks, j <= i + WIN.
__global__ __launch_bounds__(128)
void attn_kernel(const bf16_t* __restrict__ Q, const bf16_t* __restrict__ K,
                 const bf16_t* __restrict__ VT, bf16_t* __restrict__ Out) {
  __shared__ __align__(16) bf16_t lds_p[4 * 16 * 32];   // per-wave 16x32 P tile
  const int lane = threadIdx.x & 31;
  const int wv   = threadIdx.x >> 5;
  const int h    = lane >> 4;
  const int ln   = lane & 15;
  const int b    = blockIdx.z;
  const int hh   = blockIdx.y;
  const int i0   = (blockIdx.x * 4 + wv) * 16;
  const size_t bh = (size_t)(b * NHEAD + hh);
  const bf16_t* q  = Q  + bh * (size_t)SEQ * HDIM;
  const bf16_t* k  = K  + bh * (size_t)SEQ * HDIM;
  const bf16_t* vt = VT + bh * (size_t)HDIM * SEQ;

  ABfrag qf[4];                     // resident Q fragments, 16x128
#pragma unroll
  for (int c = 0; c < 4; ++c) {
    const bf16_t* qp = q + (size_t)(i0 + ln) * HDIM + c * 32;
    qf[c].h[0] = *(const v8bf*)(qp + 16 * h);
    qf[c].h[1] = *(const v8bf*)(qp + 16 * h + 8);
  }

  v8f o[8] = {};                    // 16x128 f32 output accumulator
  float mrow[8], lrow[8];
#pragma unroll
  for (int r = 0; r < 8; ++r) { mrow[r] = -1.0e30f; lrow[r] = 0.0f; }

  // base-2 softmax: fold 1/sqrt(128) * log2(e) into the score scale
  const float scale2 = 0.08838834764831845f * 1.4426950408889634f;
  int jmax = i0 + 15 + WIN + 1;
  if (jmax > SEQ) jmax = SEQ;

  bf16_t*       pwr = lds_p + wv * 512 + (8 * h) * 32;  // write rows 8h..8h+7
  const bf16_t* prd = lds_p + wv * 512 + ln * 32;       // read row ln (A layout)

  for (int kb = 0; kb < jmax; kb += 32) {
    v8f s0 = {}, s1 = {};
#pragma unroll
    for (int c = 0; c < 4; ++c) {
      v16bf kf0 = *(const v16bf*)(k + (size_t)(kb      + ln) * HDIM + c * 32 + 16 * h);
      v16bf kf1 = *(const v16bf*)(k + (size_t)(kb + 16 + ln) * HDIM + c * 32 + 16 * h);
      s0 = wmma_bf16(qf[c].v, kf0, s0);
      s1 = wmma_bf16(qf[c].v, kf1, s1);
    }
    const int j0 = kb + ln;
    const int j1 = kb + 16 + ln;
    float corr[8];
#pragma unroll
    for (int r = 0; r < 8; ++r) {
      const int i = i0 + 8 * h + r;
      float a0 = s0[r] * scale2;
      float a1 = s1[r] * scale2;
      if (j0 - i > WIN) a0 = -1.0e30f;
      if (j1 - i > WIN) a1 = -1.0e30f;
      float mx = fmaxf(a0, a1);                 // row-max across 16 lanes
#pragma unroll
      for (int off = 8; off > 0; off >>= 1)
        mx = fmaxf(mx, __shfl_xor(mx, off, 32));
      const float mnew = fmaxf(mrow[r], mx);
      corr[r] = exp2f(mrow[r] - mnew);
      const float p0 = exp2f(a0 - mnew);
      const float p1 = exp2f(a1 - mnew);
      float rs = p0 + p1;
#pragma unroll
      for (int off = 8; off > 0; off >>= 1)
        rs += __shfl_xor(rs, off, 32);
      lrow[r] = lrow[r] * corr[r] + rs;
      mrow[r] = mnew;
      pwr[r * 32 + ln]      = (bf16_t)p0;       // C-layout -> row-major LDS
      pwr[r * 32 + 16 + ln] = (bf16_t)p1;
    }
#pragma unroll
    for (int nt = 0; nt < 8; ++nt)
#pragma unroll
      for (int r = 0; r < 8; ++r)
        o[nt][r] *= corr[r];

    ABfrag pf;                                  // reload P in A layout
    pf.h[0] = *(const v8bf*)(prd + 16 * h);
    pf.h[1] = *(const v8bf*)(prd + 16 * h + 8);
#pragma unroll
    for (int nt = 0; nt < 8; ++nt) {
      v16bf vf = *(const v16bf*)(vt + (size_t)(nt * 16 + ln) * SEQ + kb + 16 * h);
      o[nt] = wmma_bf16(pf.v, vf, o[nt]);
    }
  }

  float inv[8];
#pragma unroll
  for (int r = 0; r < 8; ++r) inv[r] = 1.0f / lrow[r];
#pragma unroll
  for (int nt = 0; nt < 8; ++nt)
#pragma unroll
    for (int r = 0; r < 8; ++r) {
      const int i = i0 + 8 * h + r;
      Out[((size_t)(b * SEQ + i)) * DIMSZ + hh * HDIM + nt * 16 + ln] =
          (bf16_t)(o[nt][r] * inv[r]);
    }
}

// ---------------------------------------------------------------- launch
extern "C" void kernel_launch(void* const* d_in, const int* in_sizes, int n_in,
                              void* d_out, int out_size, void* d_ws, size_t ws_size,
                              hipStream_t stream) {
  (void)in_sizes; (void)n_in; (void)out_size; (void)ws_size;
  const float* x  = (const float*)d_in[0];
  const float* wq = (const float*)d_in[1];
  const float* wk = (const float*)d_in[2];
  const float* wv = (const float*)d_in[3];
  const float* wo = (const float*)d_in[4];

  const size_t nX = (size_t)NBATCH * SEQ * DIMSZ;  // 8,388,608
  const size_t nW = (size_t)DIMSZ * DIMSZ;         // 4,194,304

  bf16_t* xb  = (bf16_t*)d_ws;     // workspace layout, ~112 MB bf16 total
  bf16_t* wqb = xb  + nX;
  bf16_t* wkb = wqb + nW;
  bf16_t* wvb = wkb + nW;
  bf16_t* wob = wvb + nW;
  bf16_t* qb  = wob + nW;          // [B,H,S,D]
  bf16_t* kb  = qb  + nX;          // [B,H,S,D]
  bf16_t* vtb = kb  + nX;          // [B,H,D,S]
  bf16_t* atb = vtb + nX;          // [B*S, DIM]

  const int t = 256;
  cvt_kernel<<<(int)((nX + t - 1) / t), t, 0, stream>>>(x,  xb,  (int)nX);
  cvt_kernel<<<(int)((nW + t - 1) / t), t, 0, stream>>>(wq, wqb, (int)nW);
  cvt_kernel<<<(int)((nW + t - 1) / t), t, 0, stream>>>(wk, wkb, (int)nW);
  cvt_kernel<<<(int)((nW + t - 1) / t), t, 0, stream>>>(wv, wvb, (int)nW);
  cvt_kernel<<<(int)((nW + t - 1) / t), t, 0, stream>>>(wo, wob, (int)nW);

  dim3 gg(DIMSZ / 256, (NBATCH * SEQ) / 128);      // 8 x 32 blocks
  gemm_kernel<MODE_QK><<<gg, 256, 0, stream>>>(xb, wqb, (void*)qb);
  gemm_kernel<MODE_QK><<<gg, 256, 0, stream>>>(xb, wkb, (void*)kb);
  gemm_kernel<MODE_VT><<<gg, 256, 0, stream>>>(xb, wvb, (void*)vtb);

  dim3 ga(SEQ / 64, NHEAD, NBATCH);                // 32 x 16 x 2 blocks, 4 waves
  attn_kernel<<<ga, 128, 0, stream>>>(qb, kb, vtb, atb);

  gemm_kernel<MODE_F32><<<gg, 256, 0, stream>>>(atb, wob, d_out);
}
